// R2NN_24988119728170
// MI455X (gfx1250) — matrix operand order repaired and checked by reading, not compiled, not run on hardware
//
#include <hip/hip_runtime.h>

// CDNA5 / gfx1250: wave32, WMMA 16x16x32 bf16 with f32 accumulate.
// M=32 rows per wave: each LDS B-fragment feeds two back-to-back WMMAs.
typedef __attribute__((ext_vector_type(16))) __bf16 v16bf;
typedef __attribute__((ext_vector_type(8)))  float  v8f;

union BF16x16 {
  v16bf v;
  unsigned u[8];
};

__device__ __forceinline__ unsigned bf16pack2(float a, float b) {
#if __has_builtin(__builtin_amdgcn_cvt_pk_bf16_f32)
  typedef __attribute__((ext_vector_type(2))) __bf16 v2bf;
  union { v2bf h; unsigned u; } cv;
  cv.h = __builtin_amdgcn_cvt_pk_bf16_f32(a, b);
  return cv.u;
#else
  // single v_perm_b32: truncating f32->bf16 pack (lo = a, hi = b)
  union { float f; unsigned u; } x, y;
  x.f = a; y.f = b;
  return __builtin_amdgcn_perm(y.u, x.u, 0x07060302u);
#endif
}

__device__ __forceinline__ float fast_tanh(float x) {
#if __has_builtin(__builtin_amdgcn_tanhf)
  return __builtin_amdgcn_tanhf(x);    // v_tanh_f32 (CDNA5 trans op)
#else
  float y;
  asm("v_tanh_f32 %0, %1" : "=v"(y) : "v"(x));
  return y;
#endif
}

#define N_ROWS   262144
#define PPE_D    126
#define INPUT_D  128
#define KD       256      // 2*INPUT
#define BLOCK_T  256      // 8 wave32
#define GRID_B   512

__global__ __launch_bounds__(BLOCK_T) void r2nn_fused(
    const float* __restrict__ left,
    const float* __restrict__ right,
    const float* __restrict__ rec,
    const float* __restrict__ Ww,
    const float* __restrict__ Wb,
    const float* __restrict__ Vw,
    const float* __restrict__ Vb,
    float* __restrict__ parent,
    float* __restrict__ outv)
{
  // 64 B-fragments (c-chunk 0..7 x col-tile 0..7), each 32 lanes x 8 dwords,
  // stored in WMMA 32x16 bf16 B layout matching A: VGPR q<4 -> K=8h+2q,+1;
  // q>=4 -> K=16+8h+2(q-4),+1 (h = lane/16).
  __shared__ unsigned sW[64 * 256];   // 64 KB
  __shared__ float sV[132];           // V_w padded, shifted access sV[g+2]
  __shared__ float sB[128];           // W_b padded to 128 with zeros

  const int tid = threadIdx.x;

  // ---- Stage weights into LDS in fragment order (one-time per block) ----
  for (int i = 0; i < 64; ++i) {
    int id = i * BLOCK_T + tid;          // dword index into sW
    int v  = id & 7;                     // VGPR within fragment
    int l  = (id >> 3) & 31;             // lane within fragment
    int f  = id >> 8;                    // fragment = c*8 + j
    int c  = f >> 3;
    int j  = f & 7;
    int n  = j * 16 + (l & 15);          // output column = W_w row
    int k  = c * 32 + ((l >> 4) << 3) + ((v >> 2) << 4) + ((v & 3) << 1);  // K index
    float a = 0.f, b = 0.f;
    if (n < PPE_D) {
      a = Ww[n * KD + k];
      b = Ww[n * KD + k + 1];
    }
    sW[id] = bf16pack2(a, b);
  }
  if (tid < 132) sV[tid] = (tid < INPUT_D) ? Vw[tid] : 0.f;
  if (tid < 128) sB[tid] = (tid < PPE_D) ? Wb[tid] : 0.f;
  __syncthreads();

  const int lane   = tid & 31;
  const int wave   = tid >> 5;
  const int m      = lane & 15;              // A row / C column within tile
  const int khalf  = (lane >> 4) << 3;       // A-frag K sub-offset: 0 or 8
  const int rhalf  = (lane >> 4) << 3;       // C-frag row half: 0 or 8
  const int numTiles = N_ROWS >> 5;          // 32 rows per tile
  const int waveId   = blockIdx.x * (BLOCK_T / 32) + wave;
  const int strideW  = gridDim.x * (BLOCK_T / 32);
  const float vb = Vb[0];

  // Hoist per-lane epilogue constants out of the tile loop (LDS -> VGPR once)
  float breg[8], vreg[8];
  #pragma unroll
  for (int j = 0; j < 8; ++j) {
    const int g = j * 16 + m;
    breg[j] = sB[g];        // bias (0 for padded cols 126/127)
    vreg[j] = sV[g + 2];    // V_w shifted by rec offset (0 past col 127)
  }
  const float v0 = sV[0], v1 = sV[1];

  for (int tile = waveId; tile < numTiles; tile += strideW) {
    const int rowBase = tile << 5;
    const int arow0   = rowBase + m;         // rows 0..15 of the tile
    const int arow1   = arow0 + 16;          // rows 16..31 of the tile

    v8f acc[2][8];
    #pragma unroll
    for (int p = 0; p < 2; ++p)
      #pragma unroll
      for (int j = 0; j < 8; ++j) {
        v8f z = {0.f, 0.f, 0.f, 0.f, 0.f, 0.f, 0.f, 0.f};
        acc[p][j] = z;
      }

    const float2 rc0 = *(const float2*)(rec + arow0 * 2);
    const float2 rc1 = *(const float2*)(rec + arow1 * 2);

    // ---- K loop: 8 chunks of 32 (chunks 0-3 from left, 4-7 from right) ----
    #pragma unroll
    for (int c = 0; c < 8; ++c) {
      const float* base = (c < 4) ? left : right;
      const int    koff = (c < 4) ? c * 32 : (c - 4) * 32;
      const float* s0 = base + (size_t)arow0 * INPUT_D + koff;
      const float* s1 = base + (size_t)arow1 * INPUT_D + koff;
      // A-frag 16-bit 16x32 layout: lanes 0-15 hold K 0..7 & 16..23,
      // lanes 16-31 hold K 8..15 & 24..31 (khalf selects).
      float4 a0 = *(const float4*)(s0 + khalf);
      float4 a1 = *(const float4*)(s0 + khalf + 4);
      float4 a2 = *(const float4*)(s0 + khalf + 16);
      float4 a3 = *(const float4*)(s0 + khalf + 20);
      float4 b0 = *(const float4*)(s1 + khalf);
      float4 b1 = *(const float4*)(s1 + khalf + 4);
      float4 b2 = *(const float4*)(s1 + khalf + 16);
      float4 b3 = *(const float4*)(s1 + khalf + 20);
      BF16x16 A0, A1;
      A0.u[0] = bf16pack2(a0.x, a0.y);
      A0.u[1] = bf16pack2(a0.z, a0.w);
      A0.u[2] = bf16pack2(a1.x, a1.y);
      A0.u[3] = bf16pack2(a1.z, a1.w);
      A0.u[4] = bf16pack2(a2.x, a2.y);
      A0.u[5] = bf16pack2(a2.z, a2.w);
      A0.u[6] = bf16pack2(a3.x, a3.y);
      A0.u[7] = bf16pack2(a3.z, a3.w);
      A1.u[0] = bf16pack2(b0.x, b0.y);
      A1.u[1] = bf16pack2(b0.z, b0.w);
      A1.u[2] = bf16pack2(b1.x, b1.y);
      A1.u[3] = bf16pack2(b1.z, b1.w);
      A1.u[4] = bf16pack2(b2.x, b2.y);
      A1.u[5] = bf16pack2(b2.z, b2.w);
      A1.u[6] = bf16pack2(b3.x, b3.y);
      A1.u[7] = bf16pack2(b3.z, b3.w);

      const unsigned* wfrag = &sW[(c * 8) * 256 + lane * 8];
      #pragma unroll
      for (int j = 0; j < 8; ++j) {
        BF16x16 B;
        #pragma unroll
        for (int q = 0; q < 8; ++q) B.u[q] = wfrag[j * 256 + q];
        // One B-fragment feeds two WMMAs (rows 0-15 and 16-31)
        acc[0][j] = __builtin_amdgcn_wmma_f32_16x16x32_bf16(
            false, A0.v, false, B.v, (short)0, acc[0][j], false, false);
        acc[1][j] = __builtin_amdgcn_wmma_f32_16x16x32_bf16(
            false, A1.v, false, B.v, (short)0, acc[1][j], false, false);
      }
    }

    // ---- Epilogue: bias + tanh, parent store (shifted by 2), V dot ----
    if (lane < 16) {
      *(float2*)(parent + (size_t)arow0 * INPUT_D) = rc0;  // parent[:,0:2] = rec
      *(float2*)(parent + (size_t)arow1 * INPUT_D) = rc1;
    }
    const float recdot0 = rc0.x * v0 + rc0.y * v1 + vb;
    const float recdot1 = rc1.x * v0 + rc1.y * v1 + vb;

    #pragma unroll
    for (int p = 0; p < 2; ++p) {
      const float recdot = p ? recdot1 : recdot0;
      #pragma unroll
      for (int r = 0; r < 8; ++r) {
        const int prow = rowBase + p * 16 + rhalf + r;  // C layout rows
        // Phase 1: 8 independent trans ops (covers the trans-op hazard rule:
        // >=7 independent instructions before each result's first use).
        float th[8];
        #pragma unroll
        for (int j = 0; j < 8; ++j)
          th[j] = fast_tanh(acc[p][j][r] + breg[j]);
        // Phase 2: V-dot partial + parent stores
        float ps = 0.f;
        #pragma unroll
        for (int j = 0; j < 8; ++j) {
          const int g = j * 16 + m;            // gemm output column
          ps += th[j] * vreg[j];               // vreg is 0 for padded cols
          if (g < PPE_D)
            parent[(size_t)prow * INPUT_D + g + 2] = th[j];
        }
        // reduce across the 16 lanes of each half-wave
        ps += __shfl_xor(ps, 1);
        ps += __shfl_xor(ps, 2);
        ps += __shfl_xor(ps, 4);
        ps += __shfl_xor(ps, 8);
        float rd = __shfl(recdot, rhalf + r);  // rec part held by lane (rhalf+r)
        if (m == 0) outv[prow] = ps + rd;
      }
    }
  }
}

extern "C" void kernel_launch(void* const* d_in, const int* in_sizes, int n_in,
                              void* d_out, int out_size, void* d_ws, size_t ws_size,
                              hipStream_t stream) {
  const float* left  = (const float*)d_in[0];
  const float* right = (const float*)d_in[1];
  const float* rec   = (const float*)d_in[2];
  const float* Ww    = (const float*)d_in[3];
  const float* Wb    = (const float*)d_in[4];
  const float* Vw    = (const float*)d_in[5];
  const float* Vb    = (const float*)d_in[6];
  float* parent = (float*)d_out;
  float* outp   = parent + (size_t)N_ROWS * INPUT_D;
  r2nn_fused<<<GRID_B, BLOCK_T, 0, stream>>>(left, right, rec, Ww, Wb, Vw, Vb,
                                             parent, outp);
}
